// LEMcuda_25091198943849
// MI455X (gfx1250) — matrix operand order repaired and checked by reading, not compiled, not run on hardware
//
#include <hip/hip_runtime.h>
#include <hip/hip_bf16.h>

#define T_STEPS 1024
#define BATCH   64
#define NINP    128
#define NHID    256
#define DT_VAL  1.0f

typedef __bf16 bf16_t;
typedef __attribute__((ext_vector_type(16))) __bf16 v16bf;
typedef __attribute__((ext_vector_type(8)))  __bf16 bf16x8;
typedef __attribute__((ext_vector_type(8)))  float  v8f;

// ---- workspace layout (bytes) ----
// preX fragments: [4096 rowtiles][48 coltiles][32 lanes][8] f32
static constexpr size_t PRE_OFF = 0;
static constexpr size_t PRE_BYTES = (size_t)4096 * 48 * 32 * 8 * 4;      // 201326592
// linX fragments: [4096][16][32][8] f32
static constexpr size_t LIN_OFF = PRE_OFF + PRE_BYTES;
static constexpr size_t LIN_BYTES = (size_t)4096 * 16 * 32 * 8 * 4;      // 67108864
static constexpr size_t WB_OFF  = LIN_OFF + LIN_BYTES;                   // W  bf16 [768][384]
static constexpr size_t WB_BYTES = (size_t)768 * 384 * 2;
static constexpr size_t WZB_OFF = WB_OFF + WB_BYTES;                     // Wz bf16 [256][384]
static constexpr size_t WZB_BYTES = (size_t)256 * 384 * 2;
static constexpr size_t XB_OFF  = WZB_OFF + WZB_BYTES;                   // x  bf16 [T*B][128]
// total ~286 MB

// Build a v16bf A/B fragment from two 16-byte chunks (callers use contiguous k = 16*half + s in A and B).
__device__ __forceinline__ v16bf load_frag2(const bf16_t* p0, const bf16_t* p1) {
  v16bf f;
  ((bf16x8*)&f)[0] = *(const bf16x8*)p0;
  ((bf16x8*)&f)[1] = *(const bf16x8*)p1;
  return f;
}

__device__ __forceinline__ float fast_sigmoid(float x) {
  return 1.0f / (1.0f + __expf(-x));
}

// ---------------------------------------------------------------------------
// Kernel 1: f32 -> bf16 conversion of weights and inputs (fully parallel).
// ---------------------------------------------------------------------------
__global__ __launch_bounds__(256) void lem_convert(
    const float* __restrict__ w, const float* __restrict__ wz,
    const float* __restrict__ x,
    bf16_t* __restrict__ wb, bf16_t* __restrict__ wzb, bf16_t* __restrict__ xb) {
  size_t i = (size_t)blockIdx.x * blockDim.x + threadIdx.x;
  if (i < (size_t)768 * 384) wb[i]  = (bf16_t)w[i];
  if (i < (size_t)256 * 384) wzb[i] = (bf16_t)wz[i];
  if (i < (size_t)T_STEPS * BATCH * NINP) xb[i] = (bf16_t)x[i];
}

// ---------------------------------------------------------------------------
// Kernel 2: input projections (parallel over all T*B rows).
//   preX = x @ Wx^T + bias        -> coltiles 0..47
//   linX = x @ Wzx^T + bias_lin   -> coltiles 48..63
// Results stored fragment-linear: [rowtile][coltile][lane][8] f32.
// ---------------------------------------------------------------------------
__global__ __launch_bounds__(256) void lem_input_proj(
    const bf16_t* __restrict__ xb, const bf16_t* __restrict__ wb,
    const bf16_t* __restrict__ wzb, const float* __restrict__ bias,
    const float* __restrict__ biasz,
    float* __restrict__ preX, float* __restrict__ linX) {
  const int lane = threadIdx.x & 31;
  const int wv   = threadIdx.x >> 5;                 // 8 waves / block
  const long tile = (long)blockIdx.x * 8 + wv;       // 4096*64 tiles total
  const int rowtile = (int)(tile >> 6);
  const int col     = (int)(tile & 63);
  const int N = lane & 15, half = lane >> 4;
  const bool isPre = (col < 48);
  const bf16_t* W = isPre ? wb : wzb;
  const int obase = (isPre ? col : (col - 48)) * 16;
  const float bv = (isPre ? bias : biasz)[obase + N];

  v8f acc;
#pragma unroll
  for (int r = 0; r < 8; ++r) acc[r] = bv;           // bias depends on col only

  const bf16_t* aptr = xb + (size_t)(rowtile * 16 + N) * NINP;   // A row (M = lane&15)
  const bf16_t* bptr = W  + (size_t)(obase + N) * 384;           // B col = W row, contiguous K
#pragma unroll
  for (int k = 0; k < 4; ++k) {                      // K = 128 = 4 x 32
    v16bf af = load_frag2(aptr + k * 32 + half * 16, aptr + k * 32 + half * 16 + 8);
    v16bf bf = load_frag2(bptr + k * 32 + half * 16, bptr + k * 32 + half * 16 + 8);
    acc = __builtin_amdgcn_wmma_f32_16x16x32_bf16(false, af, false, bf,
                                                  (short)0, acc, false, false);
  }
  float* dst = isPre
      ? preX + (((size_t)rowtile * 48 + col) * 32 + lane) * 8
      : linX + (((size_t)rowtile * 16 + (col - 48)) * 32 + lane) * 8;
  *(v8f*)dst = acc;
}

// ---------------------------------------------------------------------------
// Kernel 3: the serial scan. One workgroup (1 WGP), 32 waves, persistent
// over all 1024 steps. State: bf16 copies in LDS for WMMA A-fragments;
// f32 state of record is the output buffer itself (L2-resident).
// Wave wv: column slice j = wv>>1 (16 hidden cols), row tiles mbase..mbase+1.
// ---------------------------------------------------------------------------
__global__ __launch_bounds__(1024) void lem_scan(
    const float* __restrict__ y0, const float* __restrict__ z0,
    const float* __restrict__ preX, const float* __restrict__ linX,
    const bf16_t* __restrict__ wb, const bf16_t* __restrict__ wzb,
    float* __restrict__ out_y, float* __restrict__ out_z) {
  __shared__ bf16_t sY[BATCH * NHID];   // 32 KB
  __shared__ bf16_t sZ[BATCH * NHID];   // 32 KB

  const int tid = threadIdx.x;
  for (int i = tid; i < BATCH * NHID; i += 1024) {
    sY[i] = (bf16_t)y0[i];
    sZ[i] = (bf16_t)z0[i];
  }
  __syncthreads();

  const int lane  = tid & 31;
  const int wv    = tid >> 5;            // 0..31
  const int j     = wv >> 1;             // hidden-col slice 0..15
  const int mbase = (wv & 1) * 2;        // row-tile base: 0 or 2
  const int N = lane & 15, half = lane >> 4;
  const int hcol = j * 16 + N;           // hidden index of C/D elements
  const int brow0 = mbase * 16 + half * 8;

  // B-fragment row pointers (constant across steps; recurrent K starts at col 128)
  const bf16_t* b1ptr[3];
#pragma unroll
  for (int g = 0; g < 3; ++g)
    b1ptr[g] = wb + (size_t)(g * 256 + j * 16 + N) * 384 + 128 + half * 16;
  const bf16_t* b2ptr = wzb + (size_t)(j * 16 + N) * 384 + 128 + half * 16;

  // A-fragment row pointers into LDS state
  const bf16_t *aY[2], *aZ[2];
#pragma unroll
  for (int m2 = 0; m2 < 2; ++m2) {
    int arow = (mbase + m2) * 16 + N;
    aY[m2] = sY + arow * NHID + half * 16;
    aZ[m2] = sZ + arow * NHID + half * 16;
  }

  float ms2r[2][8];   // dt*sigmoid(g2), held in VGPRs across the mid-step barrier

  for (int t = 0; t < T_STEPS; ++t) {
    // -------- GEMM1: pre = y @ Wh^T + preX[t] (K = 256) --------
    v8f acc[2][3];
#pragma unroll
    for (int m2 = 0; m2 < 2; ++m2) {
      size_t rt = (size_t)(t * 4 + mbase + m2);
#pragma unroll
      for (int g = 0; g < 3; ++g)
        acc[m2][g] = *(const v8f*)(preX + ((rt * 48 + g * 16 + j) * 32 + lane) * 8);
    }
    if (t + 1 < T_STEPS)   // warm L2/WGP$ for next step's fragments
      __builtin_prefetch(preX + (((size_t)((t + 1) * 4 + mbase) * 48 + j) * 32 + lane) * 8, 0, 0);

#pragma unroll
    for (int k = 0; k < 8; ++k) {
      v16bf bfrag[3];
#pragma unroll
      for (int g = 0; g < 3; ++g)
        bfrag[g] = load_frag2(b1ptr[g] + k * 32, b1ptr[g] + k * 32 + 8);
      v16bf afrag[2];
#pragma unroll
      for (int m2 = 0; m2 < 2; ++m2)
        afrag[m2] = load_frag2(aY[m2] + k * 32, aY[m2] + k * 32 + 8);
#pragma unroll
      for (int m2 = 0; m2 < 2; ++m2)
#pragma unroll
        for (int g = 0; g < 3; ++g)
          acc[m2][g] = __builtin_amdgcn_wmma_f32_16x16x32_bf16(
              false, afrag[m2], false, bfrag[g], (short)0, acc[m2][g], false, false);
    }

    // z update (wave owns matching g1/g2/a columns -> fully in-register)
#pragma unroll
    for (int m2 = 0; m2 < 2; ++m2) {
#pragma unroll
      for (int r = 0; r < 8; ++r) {
        int b = brow0 + m2 * 16 + r;
        float ms1 = DT_VAL * fast_sigmoid(acc[m2][0][r]);
        ms2r[m2][r] = DT_VAL * fast_sigmoid(acc[m2][1][r]);
        float zo = (t == 0) ? z0[b * NHID + hcol]
                            : out_z[((size_t)(t - 1) * BATCH + b) * NHID + hcol];
        float zn = (1.0f - ms1) * zo + ms1 * tanhf(acc[m2][2][r]);
        out_z[((size_t)t * BATCH + b) * NHID + hcol] = zn;
        sZ[b * NHID + hcol] = (bf16_t)zn;
      }
    }
    __syncthreads();

    // -------- GEMM2: lin = z_new @ Wzh^T + linX[t] (K = 256) --------
    v8f acc2[2];
#pragma unroll
    for (int m2 = 0; m2 < 2; ++m2) {
      size_t rt = (size_t)(t * 4 + mbase + m2);
      acc2[m2] = *(const v8f*)(linX + ((rt * 16 + j) * 32 + lane) * 8);
    }
#pragma unroll
    for (int k = 0; k < 8; ++k) {
      v16bf bfrag = load_frag2(b2ptr + k * 32, b2ptr + k * 32 + 8);
#pragma unroll
      for (int m2 = 0; m2 < 2; ++m2) {
        v16bf afrag = load_frag2(aZ[m2] + k * 32, aZ[m2] + k * 32 + 8);
        acc2[m2] = __builtin_amdgcn_wmma_f32_16x16x32_bf16(
            false, afrag, false, bfrag, (short)0, acc2[m2], false, false);
      }
    }

    // y update
#pragma unroll
    for (int m2 = 0; m2 < 2; ++m2) {
#pragma unroll
      for (int r = 0; r < 8; ++r) {
        int b = brow0 + m2 * 16 + r;
        float yo = (t == 0) ? y0[b * NHID + hcol]
                            : out_y[((size_t)(t - 1) * BATCH + b) * NHID + hcol];
        float ms2 = ms2r[m2][r];
        float yn = (1.0f - ms2) * yo + ms2 * tanhf(acc2[m2][r]);
        out_y[((size_t)t * BATCH + b) * NHID + hcol] = yn;
        sY[b * NHID + hcol] = (bf16_t)yn;
      }
    }
    __syncthreads();   // protects sY/sZ reuse in next step's GEMM1
  }
}

// ---------------------------------------------------------------------------
extern "C" void kernel_launch(void* const* d_in, const int* in_sizes, int n_in,
                              void* d_out, int out_size, void* d_ws, size_t ws_size,
                              hipStream_t stream) {
  (void)in_sizes; (void)n_in; (void)out_size; (void)ws_size;
  const float* x  = (const float*)d_in[0];   // [T,B,128]
  const float* w  = (const float*)d_in[1];   // [768,384]
  const float* wz = (const float*)d_in[2];   // [256,384]
  const float* b  = (const float*)d_in[3];   // [768]
  const float* bz = (const float*)d_in[4];   // [256]
  const float* y0 = (const float*)d_in[5];   // [64,256]
  const float* z0 = (const float*)d_in[6];   // [64,256]

  float* out_y = (float*)d_out;                                    // all_y [T,B,256]
  float* out_z = out_y + (size_t)T_STEPS * BATCH * NHID;           // all_z [T,B,256]

  char* ws = (char*)d_ws;
  float*  preX = (float*)(ws + PRE_OFF);
  float*  linX = (float*)(ws + LIN_OFF);
  bf16_t* wb   = (bf16_t*)(ws + WB_OFF);
  bf16_t* wzb  = (bf16_t*)(ws + WZB_OFF);
  bf16_t* xb   = (bf16_t*)(ws + XB_OFF);

  // Phase 0: bf16 packs (8.4M threads)
  lem_convert<<<32768, 256, 0, stream>>>(w, wz, x, wb, wzb, xb);
  // Phase 1: device-wide input projections (262144 tiles, 8 waves/block)
  lem_input_proj<<<32768, 256, 0, stream>>>(xb, wb, wzb, b, bz, preX, linX);
  // Phase 2: serial scan on one WGP
  lem_scan<<<1, 1024, 0, stream>>>(y0, z0, preX, linX, wb, wzb, out_y, out_z);
}
